// HGTEncoder_71047349010445
// MI455X (gfx1250) — matrix operand order repaired and checked by reading, not compiled, not run on hardware
//
#include <hip/hip_runtime.h>
#include <hip/hip_bf16.h>
#include <math.h>

typedef __attribute__((ext_vector_type(16))) __bf16 v16bf;
typedef __attribute__((ext_vector_type(8)))  float  v8f;
typedef __attribute__((ext_vector_type(4)))  unsigned v4u;
typedef __attribute__((ext_vector_type(8)))  int      v8i;
typedef __attribute__((ext_vector_type(4)))  int      v4i;

#define N_USER 50000
#define N_POST 100000
#define N_EDGE 300000

// ---------------- helpers ----------------
__device__ __forceinline__ unsigned short f2bf(float f) {
    unsigned u = __float_as_uint(f);
    unsigned r = u + 0x7FFFu + ((u >> 16) & 1u);   // round-to-nearest-even
    return (unsigned short)(r >> 16);
}
__device__ __forceinline__ float gelu_tanh(float x) {
    float x3 = x * x * x;
    return 0.5f * x * (1.0f + tanhf(0.7978845608028654f * (x + 0.044715f * x3)));
}
// order-preserving f32 <-> u32 for atomic max
__device__ __forceinline__ unsigned enc_f32(float f) {
    unsigned b = __float_as_uint(f);
    return (b & 0x80000000u) ? ~b : (b | 0x80000000u);
}
__device__ __forceinline__ float dec_f32(unsigned k) {
    unsigned b = (k & 0x80000000u) ? (k & 0x7FFFFFFFu) : ~k;
    return __uint_as_float(b);
}

union FragAB { v16bf v; unsigned u[8]; };

// ---------------- generic fill ----------------
__global__ void fill_u32(unsigned* p, unsigned val, long n) {
    long i = (long)blockIdx.x * blockDim.x + threadIdx.x;
    long s = (long)gridDim.x * blockDim.x;
    for (; i < n; i += s) p[i] = val;
}

// ---------------- weight pre-conversion: WT[n][k] = bf16(W[k][n]), N = 128 ----------------
__global__ void conv_wt(const float* __restrict__ W, int K, unsigned short* __restrict__ WT) {
    long i = (long)blockIdx.x * blockDim.x + threadIdx.x;
    long n = (long)K * 128;
    if (i >= n) return;
    int k = (int)(i >> 7), c = (int)(i & 127);
    WT[(long)c * K + k] = f2bf(W[i]);
}

// ---------------- GEMM: C[Mx128] = epi( bf16(A[MxK]) @ WT^T + bias ) ----------------
// WT: [128][K] bf16 (pre-transposed). 64 rows x 128 cols per block, 8 waves.
// TDM streams the 128x32 weight tile into LDS (double buffered, padded pitch 34 ush).
// MODE 0: plain store. MODE 1: A -> gelu(A) on load; out = gate*(acc+b) + (1-gate)*hold
template <int MODE>
__global__ void gemm64_tdm(const float* __restrict__ A, int lda, int M, int K,
                           const unsigned short* __restrict__ WT,
                           const float* __restrict__ bias,
                           float* __restrict__ C,
                           const float* __restrict__ hold,
                           const float* __restrict__ skipp) {
    __shared__ unsigned short sA[2][64][34];
    __shared__ unsigned short sW[2][128][34];
    const int tid  = threadIdx.x;
    const int wave = tid >> 5;
    const int lane = tid & 31;
    const int half = lane >> 4;
    const int l16  = lane & 15;
    const int rt   = wave & 3;        // row tile 0..3  (16 rows each)
    const int ch   = wave >> 2;       // col half 0..1  (64 cols each)
    const long m0  = (long)blockIdx.x * 64;
    const int nst  = K >> 5;

    v8f acc[4] = {};

    auto loadA = [&](int buf, int k0) {
        for (int i = tid; i < 64 * 32; i += 256) {
            int rr = i >> 5, kk = i & 31;
            long row = m0 + rr; if (row >= M) row = M - 1;   // clamp (stores guarded)
            float a = A[row * lda + k0 + kk];
            if (MODE == 1) a = gelu_tanh(a);
            sA[buf][rr][kk] = f2bf(a);
        }
    };
    auto tdm_issue = [&](int buf, int k0) {
        if (wave == 0) {
            unsigned lds = (unsigned)(unsigned long long)(void*)&sW[buf][0][0];
            unsigned long long ga = (unsigned long long)(const void*)(WT + k0); // tile start (n=0,k=k0)
            v4u g0 = { 1u,                                   // count=1, user mode
                       lds,                                  // lds_addr
                       (unsigned)(ga & 0xFFFFFFFFu),
                       (unsigned)((ga >> 32) & 0x01FFFFFFu) | 0x80000000u }; // addr hi | type=2
            long s1 = (long)K * 128;
            v8i g1;
            g1[0] = (1 << 16)      // data_size = 2 bytes
                  | (1 << 20)      // pad_enable
                  | (3 << 22);     // pad_interval: 16 dwords (one 32-elem bf16 row)
                                   // pad_amount code 0 -> 1 dword => pitch 34 ushorts
            g1[1] = (int)((unsigned)(K & 0xFFFF) << 16);            // tensor_dim0 lo16
            g1[2] = (int)(((unsigned)K >> 16) | (128u << 16));      // tensor_dim0 hi | tensor_dim1 lo
            g1[3] = (int)(32u << 16);                               // tensor_dim1 hi(0) | tile_dim0 = 32
            g1[4] = 128;                                            // tile_dim1 = 128 | tile_dim2 = 0
            g1[5] = (int)(unsigned)K;                               // tensor_dim0_stride lo32
            g1[6] = (int)(((unsigned)(s1 & 0xFFFF)) << 16);         // stride0 hi(0) | stride1 lo16
            g1[7] = (int)((unsigned long long)s1 >> 16);            // stride1 hi
            v4i gz = { 0, 0, 0, 0 };
#if defined(__clang_major__) && __clang_major__ >= 23
            v8i z8 = { 0, 0, 0, 0, 0, 0, 0, 0 };
            __builtin_amdgcn_tensor_load_to_lds(g0, g1, gz, gz, z8, 0);
#else
            __builtin_amdgcn_tensor_load_to_lds(g0, g1, gz, gz, 0);
#endif
        }
    };

    // prologue: stage step 0
    tdm_issue(0, 0);
    loadA(0, 0);
    if (wave == 0) __builtin_amdgcn_s_wait_tensorcnt(0);
    __syncthreads();

    for (int s = 0; s < nst; ++s) {
        const int cur = s & 1, nxt = cur ^ 1;
        if (s + 1 < nst) {
            tdm_issue(nxt, (s + 1) << 5);
            loadA(nxt, (s + 1) << 5);
        }
        FragAB fa;
#pragma unroll
        for (int v = 0; v < 8; ++v) {
            int ka = 16 * half + 2 * v;  // A 16x32 bf16 layout
            fa.u[v] = *(const unsigned*)&sA[cur][rt * 16 + l16][ka];
        }
#pragma unroll
        for (int t = 0; t < 4; ++t) {
            FragAB fb;
            const int c = ch * 64 + t * 16 + l16;
#pragma unroll
            for (int v = 0; v < 8; ++v)                              // B 32x16 bf16 layout
                fb.u[v] = *(const unsigned*)&sW[cur][c][16 * half + 2 * v];
            acc[t] = __builtin_amdgcn_wmma_f32_16x16x32_bf16(false, fa.v, false, fb.v,
                                                             (short)0, acc[t], false, false);
        }
        if (s + 1 < nst && wave == 0) __builtin_amdgcn_s_wait_tensorcnt(0);
        __syncthreads();
    }

    float gate = 0.f;
    if (MODE == 1) gate = 1.0f / (1.0f + expf(-skipp[0]));
#pragma unroll
    for (int t = 0; t < 4; ++t) {
        const int c = ch * 64 + t * 16 + l16;
#pragma unroll
        for (int r = 0; r < 8; ++r) {
            long row = m0 + rt * 16 + r + 8 * half;   // C layout: VGPR r -> M = r + 8*half
            if (row < M) {
                float o = acc[t][r] + bias[c];
                if (MODE == 1) o = gate * o + (1.0f - gate) * hold[row * 128 + c];
                C[row * 128 + c] = o;
            }
        }
    }
}

// ---------------- edge logits + segment max ----------------
// block = 128 thr (4 waves = 4 heads), 16 edges per block
__global__ void edge_logit(const float* __restrict__ kbuf,
                           const float* __restrict__ qbuf,
                           const float* __restrict__ a_rel,  // [4][32][32]
                           const float* __restrict__ p_rel,  // [4]
                           const int* __restrict__ src, const int* __restrict__ dst,
                           float* __restrict__ logit,        // [E][4]
                           unsigned* __restrict__ mx) {      // [Ndst][4] encoded
    __shared__ int s_src[16], s_dst[16];
    __shared__ unsigned short sK[4][16][34];
    __shared__ unsigned short sBT[4][32][34];
    const int tid = threadIdx.x;
    const int h = tid >> 5, lane = tid & 31;
    const int e0 = blockIdx.x << 4;
    if (tid < 16) s_src[tid] = src[e0 + tid];
    else if (tid < 32) s_dst[tid - 16] = dst[e0 + tid - 16];
    __syncthreads();
    for (int i = lane; i < 16 * 32; i += 32) {
        int e = i >> 5, d = i & 31;
        sK[h][e][d] = f2bf(kbuf[(long)s_src[e] * 128 + h * 32 + d]);
    }
    for (int i = lane; i < 32 * 32; i += 32) {
        int d = i >> 5, f = i & 31;
        sBT[h][f][d] = f2bf(a_rel[h * 1024 + i]);   // i = d*32+f
    }
    __syncthreads();
    const int half = lane >> 4, l16 = lane & 15;
    FragAB fa, fb0, fb1;
#pragma unroll
    for (int v = 0; v < 8; ++v) {
        int ka = 16 * half + 2 * v;
        fa.u[v] = *(const unsigned*)&sK[h][l16][ka];
        int kb = 16 * half + 2 * v;
        fb0.u[v] = *(const unsigned*)&sBT[h][l16][kb];
        fb1.u[v] = *(const unsigned*)&sBT[h][l16 + 16][kb];
    }
    v8f acc0 = {}, acc1 = {};
    acc0 = __builtin_amdgcn_wmma_f32_16x16x32_bf16(false, fa.v, false, fb0.v, (short)0, acc0, false, false);
    acc1 = __builtin_amdgcn_wmma_f32_16x16x32_bf16(false, fa.v, false, fb1.v, (short)0, acc1, false, false);
    const float scale = p_rel[h] * 0.17677669529663687f;  // 1/sqrt(32)
#pragma unroll
    for (int r = 0; r < 8; ++r) {
        int e = r + 8 * half;
        const float* qrow = qbuf + (long)s_dst[e] * 128 + h * 32;
        float part = acc0[r] * qrow[l16] + acc1[r] * qrow[l16 + 16];
        part += __shfl_xor(part, 1);
        part += __shfl_xor(part, 2);
        part += __shfl_xor(part, 4);
        part += __shfl_xor(part, 8);
        if (l16 == 0) {
            float lg = part * scale;
            logit[(long)(e0 + e) * 4 + h] = lg;
            atomicMax(&mx[(long)s_dst[e] * 4 + h], enc_f32(lg));
        }
    }
}

// ---------------- ex = exp(logit - mx[dst]); den += ex ----------------
__global__ void ex_den(float* __restrict__ lg, const unsigned* __restrict__ mx,
                       const int* __restrict__ dst, float* __restrict__ den, int E) {
    int e = blockIdx.x * blockDim.x + threadIdx.x;
    if (e >= E) return;
    int d = dst[e];
#pragma unroll
    for (int h = 0; h < 4; ++h) {
        float m = dec_f32(mx[(long)d * 4 + h]);
        float ex = expf(lg[(long)e * 4 + h] - m);
        lg[(long)e * 4 + h] = ex;
        atomicAdd(&den[(long)d * 4 + h], ex);
    }
}

// ---------------- msg = v[src]@m_rel ; agg[dst] += alpha*msg ----------------
__global__ void edge_agg(const float* __restrict__ vbuf,
                         const float* __restrict__ m_rel,
                         const int* __restrict__ src, const int* __restrict__ dst,
                         const float* __restrict__ ex, const float* __restrict__ den,
                         float* __restrict__ agg) {
    __shared__ int s_src[16], s_dst[16];
    __shared__ unsigned short sV[4][16][34];
    __shared__ unsigned short sBT[4][32][34];
    const int tid = threadIdx.x;
    const int h = tid >> 5, lane = tid & 31;
    const int e0 = blockIdx.x << 4;
    if (tid < 16) s_src[tid] = src[e0 + tid];
    else if (tid < 32) s_dst[tid - 16] = dst[e0 + tid - 16];
    __syncthreads();
    for (int i = lane; i < 16 * 32; i += 32) {
        int e = i >> 5, d = i & 31;
        sV[h][e][d] = f2bf(vbuf[(long)s_src[e] * 128 + h * 32 + d]);
    }
    for (int i = lane; i < 32 * 32; i += 32) {
        int d = i >> 5, f = i & 31;
        sBT[h][f][d] = f2bf(m_rel[h * 1024 + i]);
    }
    __syncthreads();
    const int half = lane >> 4, l16 = lane & 15;
    FragAB fa, fb0, fb1;
#pragma unroll
    for (int v = 0; v < 8; ++v) {
        int ka = 16 * half + 2 * v;
        fa.u[v] = *(const unsigned*)&sV[h][l16][ka];
        int kb = 16 * half + 2 * v;
        fb0.u[v] = *(const unsigned*)&sBT[h][l16][kb];
        fb1.u[v] = *(const unsigned*)&sBT[h][l16 + 16][kb];
    }
    v8f acc0 = {}, acc1 = {};
    acc0 = __builtin_amdgcn_wmma_f32_16x16x32_bf16(false, fa.v, false, fb0.v, (short)0, acc0, false, false);
    acc1 = __builtin_amdgcn_wmma_f32_16x16x32_bf16(false, fa.v, false, fb1.v, (short)0, acc1, false, false);
#pragma unroll
    for (int r = 0; r < 8; ++r) {
        int e = r + 8 * half;
        int gd = s_dst[e];
        float alpha = ex[(long)(e0 + e) * 4 + h] / den[(long)gd * 4 + h];
        atomicAdd(&agg[(long)gd * 128 + h * 32 + l16],      acc0[r] * alpha);
        atomicAdd(&agg[(long)gd * 128 + h * 32 + l16 + 16], acc1[r] * alpha);
    }
}

// ---------------- out = LN(hin + nh) ; one wave per 128-wide row ----------------
__global__ void residual_ln(const float* __restrict__ hin, const float* __restrict__ nh,
                            const float* __restrict__ g, const float* __restrict__ b,
                            float* __restrict__ out) {
    const int lane = threadIdx.x & 31;
    const long row = (long)blockIdx.x * 4 + (threadIdx.x >> 5);
    float x[4];
    float s = 0.f;
#pragma unroll
    for (int j = 0; j < 4; ++j) {
        int c = lane + 32 * j;
        x[j] = hin[row * 128 + c] + nh[row * 128 + c];
        s += x[j];
    }
#pragma unroll
    for (int o = 16; o >= 1; o >>= 1) s += __shfl_xor(s, o);
    float mu = s * (1.0f / 128.0f);
    float vs = 0.f;
#pragma unroll
    for (int j = 0; j < 4; ++j) { float d = x[j] - mu; vs += d * d; }
#pragma unroll
    for (int o = 16; o >= 1; o >>= 1) vs += __shfl_xor(vs, o);
    float inv = rsqrtf(vs * (1.0f / 128.0f) + 1e-5f);
#pragma unroll
    for (int j = 0; j < 4; ++j) {
        int c = lane + 32 * j;
        out[row * 128 + c] = (x[j] - mu) * inv * g[c] + b[c];
    }
}

// ================= host orchestration =================
extern "C" void kernel_launch(void* const* d_in, const int* in_sizes, int n_in,
                              void* d_out, int out_size, void* d_ws, size_t ws_size,
                              hipStream_t stream) {
    (void)in_sizes; (void)n_in; (void)out_size; (void)ws_size;
    const float* params = (const float*)d_in[0];
    const float* x_user = (const float*)d_in[1];
    const float* x_post = (const float*)d_in[2];
    const int* src_up = (const int*)d_in[3];
    const int* dst_up = (const int*)d_in[4];
    const int* src_pu = (const int*)d_in[5];
    const int* dst_pu = (const int*)d_in[6];

    // ---- param offsets: jax tree_flatten (dict keys sorted alphabetically) ----
    size_t off = 0;
    auto take = [&](size_t n) { size_t o = off; off += n; return o; };
    const size_t o_inpost_W = take(768 * 128), o_inpost_b = take(128);
    const size_t o_inuser_W = take(64 * 128),  o_inuser_b = take(128);
    struct LO {
        size_t pu_a, pu_m, pu_p, up_a, up_m, up_p;
        size_t ln_post_b, ln_post_g, ln_user_b, ln_user_g;
        size_t post_Wk, post_Wo, post_Wq, post_Wv, post_bk, post_bo, post_bq, post_bv, post_skip;
        size_t user_Wk, user_Wo, user_Wq, user_Wv, user_bk, user_bo, user_bq, user_bv, user_skip;
    } L[2];
    for (int l = 0; l < 2; ++l) {
        L[l].pu_a = take(4096); L[l].pu_m = take(4096); L[l].pu_p = take(4);
        L[l].up_a = take(4096); L[l].up_m = take(4096); L[l].up_p = take(4);
        L[l].ln_post_b = take(128); L[l].ln_post_g = take(128);
        L[l].ln_user_b = take(128); L[l].ln_user_g = take(128);
        L[l].post_Wk = take(16384); L[l].post_Wo = take(16384);
        L[l].post_Wq = take(16384); L[l].post_Wv = take(16384);
        L[l].post_bk = take(128); L[l].post_bo = take(128);
        L[l].post_bq = take(128); L[l].post_bv = take(128);
        L[l].post_skip = take(1);
        L[l].user_Wk = take(16384); L[l].user_Wo = take(16384);
        L[l].user_Wq = take(16384); L[l].user_Wv = take(16384);
        L[l].user_bk = take(128); L[l].user_bo = take(128);
        L[l].user_bq = take(128); L[l].user_bv = take(128);
        L[l].user_skip = take(1);
    }

    // ---- workspace layout (fp32 elements) ----
    float* ws = (float*)d_ws;
    size_t w = 0;
    auto wt = [&](size_t n) { size_t o = w; w += (n + 63) & ~(size_t)63; return o; };
    float* h_user    = ws + wt((size_t)N_USER * 128);
    float* h_post    = ws + wt((size_t)N_POST * 128);
    float* k_user    = ws + wt((size_t)N_USER * 128);
    float* q_user    = ws + wt((size_t)N_USER * 128);
    float* v_user    = ws + wt((size_t)N_USER * 128);
    float* k_post    = ws + wt((size_t)N_POST * 128);
    float* q_post    = ws + wt((size_t)N_POST * 128);
    float* v_post    = ws + wt((size_t)N_POST * 128);
    float* newh_user = ws + wt((size_t)N_USER * 128);
    float* newh_post = ws + wt((size_t)N_POST * 128);
    float* logit     = ws + wt((size_t)N_EDGE * 4);
    unsigned* mx     = (unsigned*)(ws + wt((size_t)N_POST * 4));
    float* den       = ws + wt((size_t)N_POST * 4);
    float* agg       = ws + wt((size_t)N_POST * 128);
    unsigned short* wt_pool = (unsigned short*)(ws + wt(200000)); // 400000 ush >= needed

    size_t uo = 0;
    auto ut = [&](size_t n) { size_t o = uo; uo += n; return o; };
    unsigned short* wt_in_user = wt_pool + ut(128 * 64);
    unsigned short* wt_in_post = wt_pool + ut(128 * 768);
    struct WTL { unsigned short *uk, *uq, *uv, *uo, *pk, *pq, *pv, *po; } WL[2];
    for (int l = 0; l < 2; ++l) {
        WL[l].uk = wt_pool + ut(16384); WL[l].uq = wt_pool + ut(16384);
        WL[l].uv = wt_pool + ut(16384); WL[l].uo = wt_pool + ut(16384);
        WL[l].pk = wt_pool + ut(16384); WL[l].pq = wt_pool + ut(16384);
        WL[l].pv = wt_pool + ut(16384); WL[l].po = wt_pool + ut(16384);
    }

    // ---- one-time weight conversion to bf16, transposed [128][K] ----
    conv_wt<<<(64 * 128 + 255) / 256, 256, 0, stream>>>(params + o_inuser_W, 64, wt_in_user);
    conv_wt<<<(768 * 128 + 255) / 256, 256, 0, stream>>>(params + o_inpost_W, 768, wt_in_post);
    for (int l = 0; l < 2; ++l) {
        const int g = (128 * 128 + 255) / 256;
        conv_wt<<<g, 256, 0, stream>>>(params + L[l].user_Wk, 128, WL[l].uk);
        conv_wt<<<g, 256, 0, stream>>>(params + L[l].user_Wq, 128, WL[l].uq);
        conv_wt<<<g, 256, 0, stream>>>(params + L[l].user_Wv, 128, WL[l].uv);
        conv_wt<<<g, 256, 0, stream>>>(params + L[l].user_Wo, 128, WL[l].uo);
        conv_wt<<<g, 256, 0, stream>>>(params + L[l].post_Wk, 128, WL[l].pk);
        conv_wt<<<g, 256, 0, stream>>>(params + L[l].post_Wq, 128, WL[l].pq);
        conv_wt<<<g, 256, 0, stream>>>(params + L[l].post_Wv, 128, WL[l].pv);
        conv_wt<<<g, 256, 0, stream>>>(params + L[l].post_Wo, 128, WL[l].po);
    }

    const unsigned ENC_NEG_INF = 0x007FFFFFu;
    const int GU = (N_USER + 63) / 64, GP = (N_POST + 63) / 64;

    // ---- input projections ----
    gemm64_tdm<0><<<GU, 256, 0, stream>>>(x_user, 64, N_USER, 64, wt_in_user,
                                          params + o_inuser_b, h_user, nullptr, nullptr);
    gemm64_tdm<0><<<GP, 256, 0, stream>>>(x_post, 768, N_POST, 768, wt_in_post,
                                          params + o_inpost_b, h_post, nullptr, nullptr);

    for (int l = 0; l < 2; ++l) {
        const LO& P = L[l];
        const WTL& Q = WL[l];
        // K/Q/V projections
        gemm64_tdm<0><<<GU, 256, 0, stream>>>(h_user, 128, N_USER, 128, Q.uk, params + P.user_bk, k_user, nullptr, nullptr);
        gemm64_tdm<0><<<GU, 256, 0, stream>>>(h_user, 128, N_USER, 128, Q.uq, params + P.user_bq, q_user, nullptr, nullptr);
        gemm64_tdm<0><<<GU, 256, 0, stream>>>(h_user, 128, N_USER, 128, Q.uv, params + P.user_bv, v_user, nullptr, nullptr);
        gemm64_tdm<0><<<GP, 256, 0, stream>>>(h_post, 128, N_POST, 128, Q.pk, params + P.post_bk, k_post, nullptr, nullptr);
        gemm64_tdm<0><<<GP, 256, 0, stream>>>(h_post, 128, N_POST, 128, Q.pq, params + P.post_bq, q_post, nullptr, nullptr);
        gemm64_tdm<0><<<GP, 256, 0, stream>>>(h_post, 128, N_POST, 128, Q.pv, params + P.post_bv, v_post, nullptr, nullptr);

        // ---- dst = post, edges user__authors__post ----
        fill_u32<<<512, 256, 0, stream>>>(mx, ENC_NEG_INF, (long)N_POST * 4);
        fill_u32<<<512, 256, 0, stream>>>((unsigned*)den, 0u, (long)N_POST * 4);
        fill_u32<<<2048, 256, 0, stream>>>((unsigned*)agg, 0u, (long)N_POST * 128);
        edge_logit<<<N_EDGE / 16, 128, 0, stream>>>(k_user, q_post, params + P.up_a, params + P.up_p,
                                                    src_up, dst_up, logit, mx);
        ex_den<<<(N_EDGE + 255) / 256, 256, 0, stream>>>(logit, mx, dst_up, den, N_EDGE);
        edge_agg<<<N_EDGE / 16, 128, 0, stream>>>(v_user, params + P.up_m, src_up, dst_up, logit, den, agg);
        gemm64_tdm<1><<<GP, 256, 0, stream>>>(agg, 128, N_POST, 128, Q.po, params + P.post_bo,
                                              newh_post, h_post, params + P.post_skip);

        // ---- dst = user, edges post__by__user ----
        fill_u32<<<512, 256, 0, stream>>>(mx, ENC_NEG_INF, (long)N_USER * 4);
        fill_u32<<<512, 256, 0, stream>>>((unsigned*)den, 0u, (long)N_USER * 4);
        fill_u32<<<2048, 256, 0, stream>>>((unsigned*)agg, 0u, (long)N_USER * 128);
        edge_logit<<<N_EDGE / 16, 128, 0, stream>>>(k_post, q_user, params + P.pu_a, params + P.pu_p,
                                                    src_pu, dst_pu, logit, mx);
        ex_den<<<(N_EDGE + 255) / 256, 256, 0, stream>>>(logit, mx, dst_pu, den, N_EDGE);
        edge_agg<<<N_EDGE / 16, 128, 0, stream>>>(v_post, params + P.pu_m, src_pu, dst_pu, logit, den, agg);
        gemm64_tdm<1><<<GU, 256, 0, stream>>>(agg, 128, N_USER, 128, Q.uo, params + P.user_bo,
                                              newh_user, h_user, params + P.user_skip);

        // ---- residual + layernorm ----
        float* post_out = (l == 1) ? (float*)d_out : h_post;
        residual_ln<<<N_POST / 4, 128, 0, stream>>>(h_post, newh_post, params + P.ln_post_g, params + P.ln_post_b, post_out);
        residual_ln<<<N_USER / 4, 128, 0, stream>>>(h_user, newh_user, params + P.ln_user_g, params + P.ln_user_b, h_user);
    }
}
